// Generator_5592047419764
// MI455X (gfx1250) — matrix operand order repaired and checked by reading, not compiled, not run on hardware
//
#include <hip/hip_runtime.h>
#include <hip/hip_bf16.h>

// ---------------------------------------------------------------------------
// Generator MLP forward, CDNA5 (gfx1250), wave32.
// Split-bf16 (hi+lo limb) GEMMs on v_wmma_f32_16x16x32_bf16 (~1e-5 rel acc).
// All operands pre-converted to bf16 limbs in global memory once; GEMM staging
// is pure 16B copies (async global->LDS when available), double-buffered LDS.
// Train-mode BN absorbs b1/b2 (they cancel), so biases are dropped.
// ---------------------------------------------------------------------------

typedef __attribute__((ext_vector_type(16))) __bf16 v16bf;
typedef __attribute__((ext_vector_type(8)))  __bf16 v8bf;
typedef __attribute__((ext_vector_type(8)))  float  v8f;
typedef __attribute__((ext_vector_type(4)))  int    v4i;
typedef unsigned short ushort_t;

// async-copy pointer types: builtin expects v4i* in AS1 (global) / AS3 (LDS)
typedef __attribute__((address_space(1))) v4i gas_v4i;
typedef __attribute__((address_space(3))) v4i las_v4i;

#define BDIM   8192
#define BM     128
#define BN     64
#define BK     32
#define LDA_P  40          // padded LDS stride (bf16 elems): 80B rows, 16B aligned
#define LDB_P  40

#if defined(__has_builtin)
#if __has_builtin(__builtin_amdgcn_global_load_async_to_lds_b128) && \
    __has_builtin(__builtin_amdgcn_s_wait_asynccnt)
#define USE_ASYNC_LDS 1
#endif
#endif

__device__ __forceinline__ ushort_t f2bf_rne(float f) {
    unsigned int u = __float_as_uint(f);
    unsigned int r = u + 0x7fffu + ((u >> 16) & 1u);   // round-to-nearest-even
    return (ushort_t)(r >> 16);
}
__device__ __forceinline__ float bf2f(ushort_t b) {
    return __uint_as_float(((unsigned int)b) << 16);
}

__device__ __forceinline__ void copy16_g2l(void* lds, const void* g) {
#ifdef USE_ASYNC_LDS
    __builtin_amdgcn_global_load_async_to_lds_b128(
        (gas_v4i*)(void*)g, (las_v4i*)lds, 0, 0);
#else
    *(v8bf*)lds = *(const v8bf*)g;
#endif
}
__device__ __forceinline__ void wait_async_lds() {
#ifdef USE_ASYNC_LDS
    __builtin_amdgcn_s_wait_asynccnt(0);
#endif
}

union Frag { v16bf v; v8bf h[2]; };

// ---------------------------------------------------------------------------
// GEMM: C[M x N] = (Ahi+Alo)[M x Kp] @ (Bhi+Blo)[Kp x N], B stored transposed
// [N x Kp]. OUT_LIMB=1: write bf16 hi/lo limbs; else fp32.
// ---------------------------------------------------------------------------
template <int OUT_LIMB>
__global__ __launch_bounds__(256)
void gemm_limbs(const ushort_t* __restrict__ Ahi, const ushort_t* __restrict__ Alo,
                const ushort_t* __restrict__ Bthi, const ushort_t* __restrict__ Btlo,
                float* __restrict__ outF,
                ushort_t* __restrict__ outHi, ushort_t* __restrict__ outLo,
                int N, int Kp, int kTiles) {
    __shared__ __align__(16) ushort_t sA[2][2][BM * LDA_P];  // [buf][hi/lo]
    __shared__ __align__(16) ushort_t sB[2][2][BN * LDB_P];

    const int tid   = threadIdx.x;
    const int lane  = tid & 31;
    const int wid   = tid >> 5;
    const int wm    = (wid >> 1) * 32;
    const int wn    = (wid & 1) * 32;
    const int bm    = blockIdx.y * BM;
    const int bn    = blockIdx.x * BN;
    const int mrow  = lane & 15;
    const int khalf = lane >> 4;

    v8f acc[2][2];
#pragma unroll
    for (int i = 0; i < 2; ++i)
#pragma unroll
        for (int j = 0; j < 2; ++j) acc[i][j] = (v8f)(0.f);

    // stage tile kt into LDS buffer `buf` (pure 16B copies, no conversion)
    auto stage = [&](int kt, int buf) {
        const int k0 = kt * BK;
#pragma unroll
        for (int i = 0; i < 2; ++i) {                    // A: 128 rows x 64B
            int flat = i * 256 + tid;                    // 0..511
            int r = flat >> 2, s = flat & 3;
            long goff = (long)(bm + r) * Kp + k0 + s * 8;
            int  loff = r * LDA_P + s * 8;
            copy16_g2l(&sA[buf][0][loff], Ahi + goff);
            copy16_g2l(&sA[buf][1][loff], Alo + goff);
        }
        {                                                // B: 64 rows x 64B
            int r = tid >> 2, s = tid & 3;
            long goff = (long)(bn + r) * Kp + k0 + s * 8;
            int  loff = r * LDB_P + s * 8;
            copy16_g2l(&sB[buf][0][loff], Bthi + goff);
            copy16_g2l(&sB[buf][1][loff], Btlo + goff);
        }
    };

    stage(0, 0);
    for (int kt = 0; kt < kTiles; ++kt) {
        const int cur = kt & 1;
        wait_async_lds();
        __syncthreads();                 // buffer `cur` ready; prev reads done
        if (kt + 1 < kTiles) stage(kt + 1, cur ^ 1);   // prefetch overlaps WMMA

        Frag ahi[2], alo[2], bhi[2], blo[2];
#pragma unroll
        for (int t = 0; t < 2; ++t) {
            int row  = wm + t * 16 + mrow;
            int kOff = khalf * 8;
            ahi[t].h[0] = *(const v8bf*)&sA[cur][0][row * LDA_P + kOff];
            ahi[t].h[1] = *(const v8bf*)&sA[cur][0][row * LDA_P + kOff + 16];
            alo[t].h[0] = *(const v8bf*)&sA[cur][1][row * LDA_P + kOff];
            alo[t].h[1] = *(const v8bf*)&sA[cur][1][row * LDA_P + kOff + 16];
        }
#pragma unroll
        for (int t = 0; t < 2; ++t) {
            int col   = wn + t * 16 + mrow;
            int kbase = khalf * 8;
            bhi[t].h[0] = *(const v8bf*)&sB[cur][0][col * LDB_P + kbase];
            bhi[t].h[1] = *(const v8bf*)&sB[cur][0][col * LDB_P + kbase + 16];
            blo[t].h[0] = *(const v8bf*)&sB[cur][1][col * LDB_P + kbase];
            blo[t].h[1] = *(const v8bf*)&sB[cur][1][col * LDB_P + kbase + 16];
        }
#pragma unroll
        for (int i = 0; i < 2; ++i)
#pragma unroll
            for (int j = 0; j < 2; ++j) {
                acc[i][j] = __builtin_amdgcn_wmma_f32_16x16x32_bf16(
                    false, ahi[i].v, false, bhi[j].v, (short)0, acc[i][j], false, false);
                acc[i][j] = __builtin_amdgcn_wmma_f32_16x16x32_bf16(
                    false, ahi[i].v, false, blo[j].v, (short)0, acc[i][j], false, false);
                acc[i][j] = __builtin_amdgcn_wmma_f32_16x16x32_bf16(
                    false, alo[i].v, false, bhi[j].v, (short)0, acc[i][j], false, false);
            }
    }

    // epilogue: C/D layout (N = lane&15, M = vreg + 8*khalf)
#pragma unroll
    for (int i = 0; i < 2; ++i)
#pragma unroll
        for (int j = 0; j < 2; ++j) {
            int col   = bn + wn + j * 16 + mrow;
            int rbase = bm + wm + i * 16 + 8 * khalf;
#pragma unroll
            for (int v = 0; v < 8; ++v) {
                long idx = (long)(rbase + v) * N + col;
                float val = acc[i][j][v];
                if (OUT_LIMB) {
                    ushort_t h = f2bf_rne(val);
                    outHi[idx] = h;
                    outLo[idx] = f2bf_rne(val - bf2f(h));
                } else {
                    outF[idx] = val;
                }
            }
        }
}

// concat(x, noise) -> bf16 limbs, K zero-padded 624 -> 640
__global__ __launch_bounds__(256)
void conv_inp(const float* __restrict__ x, const float* __restrict__ noise,
              ushort_t* __restrict__ hi, ushort_t* __restrict__ lo) {
    int i = blockIdx.x * 256 + threadIdx.x;        // over 8192*640
    int r = i / 640, k = i - r * 640;
    float v = 0.f;
    if (k < 312)      v = x[r * 312 + k];
    else if (k < 624) v = noise[r * 312 + k - 312];
    ushort_t h = f2bf_rne(v);
    hi[i] = h;
    lo[i] = f2bf_rne(v - bf2f(h));
}

// W[K x N] fp32 -> transposed bf16 limbs [N x Kp], zero-padded rows (LDS-tiled)
__global__ __launch_bounds__(256)
void conv_w_transpose(const float* __restrict__ W, ushort_t* __restrict__ thi,
                      ushort_t* __restrict__ tlo, int K, int N, int Kp) {
    __shared__ ushort_t tH[64 * 33];
    __shared__ ushort_t tL[64 * 33];
    int bn = blockIdx.x * 64, bk = blockIdx.y * 32, tid = threadIdx.x;
#pragma unroll
    for (int i = 0; i < 8; ++i) {
        int flat = i * 256 + tid;
        int kk = flat >> 6, n = flat & 63;
        int gk = bk + kk;
        float v = (gk < K) ? W[(long)gk * N + bn + n] : 0.f;
        ushort_t h = f2bf_rne(v);
        tH[n * 33 + kk] = h;
        tL[n * 33 + kk] = f2bf_rne(v - bf2f(h));
    }
    __syncthreads();
#pragma unroll
    for (int i = 0; i < 8; ++i) {
        int flat = i * 256 + tid;
        int n = flat >> 5, k = flat & 31;
        long o = (long)(bn + n) * Kp + bk + k;
        thi[o] = tH[n * 33 + k];
        tlo[o] = tL[n * 33 + k];
    }
}

// In-place: limbs(raw h) -> limbs(LeakyReLU(BN1(h)) * mask)
__global__ __launch_bounds__(256)
void act_limbs(ushort_t* __restrict__ hi, ushort_t* __restrict__ lo,
               const float* __restrict__ mask, const float* __restrict__ scale,
               const float* __restrict__ shift) {
    int i = blockIdx.x * 256 + threadIdx.x;        // over 8192*4096
    int j = i & 4095;
    float v = bf2f(hi[i]) + bf2f(lo[i]);
    v = v * scale[j] + shift[j];
    v = (v >= 0.f) ? v : 0.01f * v;
    v *= mask[i];
    ushort_t h = f2bf_rne(v);
    hi[i] = h;
    lo[i] = f2bf_rne(v - bf2f(h));
}

// Column partial sums over 256-row chunks (coalesced, deterministic)
__global__ __launch_bounds__(256)
void reduce_partial_limb(const ushort_t* __restrict__ hi, const ushort_t* __restrict__ lo,
                         float* __restrict__ psum, float* __restrict__ psq, int C) {
    int j = blockIdx.x * 256 + threadIdx.x;
    int r0 = blockIdx.y * 256;
    float s = 0.f, q = 0.f;
    for (int r = 0; r < 256; ++r) {
        long idx = (long)(r0 + r) * C + j;
        float v = bf2f(hi[idx]) + bf2f(lo[idx]);
        s += v; q += v * v;
    }
    psum[blockIdx.y * C + j] = s;
    psq [blockIdx.y * C + j] = q;
}

__global__ __launch_bounds__(256)
void reduce_partial_f32(const float* __restrict__ src, float* __restrict__ psum,
                        float* __restrict__ psq, int C) {
    int j = blockIdx.x * 256 + threadIdx.x;
    int r0 = blockIdx.y * 256;
    float s = 0.f, q = 0.f;
    for (int r = 0; r < 256; ++r) {
        float v = src[(long)(r0 + r) * C + j];
        s += v; q += v * v;
    }
    psum[blockIdx.y * C + j] = s;
    psq [blockIdx.y * C + j] = q;
}

__global__ __launch_bounds__(256)
void reduce_final(const float* __restrict__ psum, const float* __restrict__ psq,
                  const float* __restrict__ g, const float* __restrict__ be,
                  float* __restrict__ scale, float* __restrict__ shift,
                  int C, int nchunks, float invB) {
    int j = blockIdx.x * 256 + threadIdx.x;
    if (j >= C) return;
    float s = 0.f, q = 0.f;
    for (int c = 0; c < nchunks; ++c) { s += psum[c * C + j]; q += psq[c * C + j]; }
    float mean = s * invB;
    float var  = q * invB - mean * mean;
    float sc   = g[j] * rsqrtf(var + 1e-5f);
    scale[j] = sc;
    shift[j] = be[j] - mean * sc;
}

__global__ __launch_bounds__(256)
void bn_relu_out(const float* __restrict__ o, const float* __restrict__ scale,
                 const float* __restrict__ shift, float* __restrict__ out, int N) {
    int idx = blockIdx.x * 256 + threadIdx.x;
    int j = idx & (N - 1);
    float v = o[idx] * scale[j] + shift[j];
    out[idx] = v > 0.f ? v : 0.f;
}

extern "C" void kernel_launch(void* const* d_in, const int* in_sizes, int n_in,
                              void* d_out, int out_size, void* d_ws, size_t ws_size,
                              hipStream_t stream) {
    const float* x     = (const float*)d_in[0];
    const float* noise = (const float*)d_in[1];
    const float* W1    = (const float*)d_in[2];
    const float* g1    = (const float*)d_in[4];   // b1 (d_in[3]) cancels in BN
    const float* be1   = (const float*)d_in[5];
    const float* W2    = (const float*)d_in[6];
    const float* g2    = (const float*)d_in[8];   // b2 (d_in[7]) cancels in BN
    const float* be2   = (const float*)d_in[9];
    const float* mask  = (const float*)d_in[10];
    float* out = (float*)d_out;

    const int HID = 4096, OUTN = 2048, KP1 = 640;

    char* p = (char*)d_ws;
    ushort_t* A2hi  = (ushort_t*)p;            p += (long)BDIM * HID * 2;   // 67MB
    ushort_t* A2lo  = (ushort_t*)p;            p += (long)BDIM * HID * 2;
    float*    o     = (float*)p;               p += (long)BDIM * OUTN * 4;  // 67MB
    ushort_t* A1hi  = (ushort_t*)p;            p += (long)BDIM * KP1 * 2;
    ushort_t* A1lo  = (ushort_t*)p;            p += (long)BDIM * KP1 * 2;
    ushort_t* W1thi = (ushort_t*)p;            p += (long)HID * KP1 * 2;
    ushort_t* W1tlo = (ushort_t*)p;            p += (long)HID * KP1 * 2;
    ushort_t* W2thi = (ushort_t*)p;            p += (long)OUTN * HID * 2;
    ushort_t* W2tlo = (ushort_t*)p;            p += (long)OUTN * HID * 2;
    float* psum1  = (float*)p;                 p += 32 * HID * 4;
    float* psq1   = (float*)p;                 p += 32 * HID * 4;
    float* scale1 = (float*)p;                 p += HID * 4;
    float* shift1 = (float*)p;                 p += HID * 4;
    float* psum2  = (float*)p;                 p += 32 * OUTN * 4;
    float* psq2   = (float*)p;                 p += 32 * OUTN * 4;
    float* scale2 = (float*)p;                 p += OUTN * 4;
    float* shift2 = (float*)p;                 p += OUTN * 4;

    dim3 blk(256);

    // one-time limb conversion / pad / transpose
    conv_inp<<<dim3((BDIM * KP1) / 256), blk, 0, stream>>>(x, noise, A1hi, A1lo);
    conv_w_transpose<<<dim3(HID / 64, KP1 / 32), blk, 0, stream>>>(W1, W1thi, W1tlo,
                                                                   624, HID, KP1);
    conv_w_transpose<<<dim3(OUTN / 64, HID / 32), blk, 0, stream>>>(W2, W2thi, W2tlo,
                                                                    HID, OUTN, HID);

    // L1 GEMM -> h limbs
    gemm_limbs<1><<<dim3(HID / BN, BDIM / BM), blk, 0, stream>>>(
        A1hi, A1lo, W1thi, W1tlo, nullptr, A2hi, A2lo, HID, KP1, KP1 / BK);

    reduce_partial_limb<<<dim3(HID / 256, BDIM / 256), blk, 0, stream>>>(A2hi, A2lo,
                                                                         psum1, psq1, HID);
    reduce_final<<<dim3(HID / 256), blk, 0, stream>>>(psum1, psq1, g1, be1, scale1,
                                                      shift1, HID, BDIM / 256, 1.f / BDIM);

    // BN1 + LeakyReLU + dropout, in place on h limbs
    act_limbs<<<dim3((BDIM * HID) / 256), blk, 0, stream>>>(A2hi, A2lo, mask,
                                                            scale1, shift1);

    // L2 GEMM -> o fp32
    gemm_limbs<0><<<dim3(OUTN / BN, BDIM / BM), blk, 0, stream>>>(
        A2hi, A2lo, W2thi, W2tlo, o, nullptr, nullptr, OUTN, HID, HID / BK);

    reduce_partial_f32<<<dim3(OUTN / 256, BDIM / 256), blk, 0, stream>>>(o, psum2,
                                                                         psq2, OUTN);
    reduce_final<<<dim3(OUTN / 256), blk, 0, stream>>>(psum2, psq2, g2, be2, scale2,
                                                       shift2, OUTN, BDIM / 256,
                                                       1.f / BDIM);

    bn_relu_out<<<dim3((BDIM * OUTN) / 256), blk, 0, stream>>>(o, scale2, shift2,
                                                               out, OUTN);
}
